// NegSimNCEHead_4440996184087
// MI455X (gfx1250) — matrix operand order repaired and checked by reading, not compiled, not run on hardware
//
#include <hip/hip_runtime.h>
#include <hip/hip_bf16.h>

#define NN 4096
#define CC 512
#define NKC  4              // K chunks of 128 per 512-K pass
#define NSTAGE (64 * NKC)   // 256 pipeline stages per gemm

typedef _Float16 v16h __attribute__((ext_vector_type(16)));
typedef _Float16 v8h  __attribute__((ext_vector_type(8)));
typedef float    v8f  __attribute__((ext_vector_type(8)));

union V16 { v16h v; v8h h[2]; };
union H8  { v8h v; _Float16 e[8]; };

__device__ __forceinline__ float waveRedAdd(float v) {
    #pragma unroll
    for (int off = 1; off < 32; off <<= 1) v += __shfl_xor(v, off, 32);
    return v;
}

// Generic shared-pointer -> LDS byte offset (flat LDS aperture keeps offset in addr[31:0]).
__device__ __forceinline__ uint32_t ldsOff(const void* p) {
    return (uint32_t)(uintptr_t)p;
}

// CDNA5 async global->LDS copy, 16 bytes per lane, GVS addressing (SGPR base + u32 VGPR offset).
__device__ __forceinline__ void asyncCopyB128(uint32_t lds_byte_off, const void* base, uint32_t g_byte_off) {
    asm volatile("global_load_async_to_lds_b128 %0, %1, %2"
                 :: "v"(lds_byte_off), "v"(g_byte_off), "s"(base) : "memory");
}
__device__ __forceinline__ void waitAsync0() {
    asm volatile("s_wait_asynccnt 0x0" ::: "memory");
}

// ---------------------------------------------------------------- zero scratch
__global__ void zero_kernel(float* p, int n) {
    int i = blockIdx.x * blockDim.x + threadIdx.x;
    if (i < n) p[i] = 0.f;
}

// ---------------------------------------------------------------- row L2-normalize -> f16, column stats
__global__ __launch_bounds__(256)
void normalize_kernel(const float* __restrict__ i0, const float* __restrict__ i1,
                      const float* __restrict__ i2, const float* __restrict__ i3,
                      _Float16* __restrict__ out16base, float* __restrict__ rnorms,
                      float* __restrict__ colsum, float* __restrict__ colsq) {
    __shared__ float ls[CC];
    __shared__ float lq[CC];
    const int mat  = blockIdx.y;
    const float* in = (mat == 0) ? i0 : (mat == 1) ? i1 : (mat == 2) ? i2 : i3;
    _Float16* out16 = out16base + (size_t)mat * NN * CC;

    const int tid  = threadIdx.x;
    const int lane = tid & 31;
    const int wave = tid >> 5;
    ls[tid] = 0.f; lq[tid] = 0.f; ls[tid + 256] = 0.f; lq[tid + 256] = 0.f;
    __syncthreads();

    const int row = blockIdx.x * 8 + wave;
    const float* rp = in + (size_t)row * CC + lane * 16;

    float x[16];
    float ssq = 0.f;
    #pragma unroll
    for (int k = 0; k < 16; k += 4) {
        float4 v = *(const float4*)(rp + k);
        x[k] = v.x; x[k+1] = v.y; x[k+2] = v.z; x[k+3] = v.w;
        ssq += v.x*v.x + v.y*v.y + v.z*v.z + v.w*v.w;
    }
    ssq = waveRedAdd(ssq);
    const float rn = 1.0f / fmaxf(sqrtf(ssq), 1e-12f);
    if (lane == 0) rnorms[mat * NN + row] = rn;

    H8 lo, hi;
    #pragma unroll
    for (int k = 0; k < 16; ++k) {
        float xn = x[k] * rn;
        if (k < 8) lo.e[k] = (_Float16)xn; else hi.e[k - 8] = (_Float16)xn;
        atomicAdd(&ls[lane * 16 + k], xn);
        atomicAdd(&lq[lane * 16 + k], xn * xn);
    }
    _Float16* op = out16 + (size_t)row * CC + lane * 16;
    *(v8h*)op       = lo.v;
    *(v8h*)(op + 8) = hi.v;

    __syncthreads();
    atomicAdd(&colsum[mat * CC + tid],        ls[tid]);
    atomicAdd(&colsum[mat * CC + tid + 256],  ls[tid + 256]);
    atomicAdd(&colsq [mat * CC + tid],        lq[tid]);
    atomicAdd(&colsq [mat * CC + tid + 256],  lq[tid + 256]);
}

// ---------------------------------------------------------------- fp32 diagonal dots
__global__ __launch_bounds__(256)
void diag_kernel(const float* __restrict__ v, const float* __restrict__ t,
                 const float* __restrict__ pv, const float* __restrict__ pt,
                 const float* __restrict__ rnorms,
                 float* __restrict__ dv, float* __restrict__ dt) {
    const int p    = blockIdx.y;
    const float* A = p ? t  : v;
    const float* B = p ? pv : pt;
    const int row  = blockIdx.x * 8 + (threadIdx.x >> 5);
    const int lane = threadIdx.x & 31;
    const float* ap = A + (size_t)row * CC + lane * 16;
    const float* bp = B + (size_t)row * CC + lane * 16;
    float d = 0.f;
    #pragma unroll
    for (int k = 0; k < 16; k += 4) {
        float4 a = *(const float4*)(ap + k);
        float4 b = *(const float4*)(bp + k);
        d += a.x*b.x + a.y*b.y + a.z*b.z + a.w*b.w;
    }
    d = waveRedAdd(d);
    if (lane == 0) {
        float rnA = rnorms[(p ? 1 : 0) * NN + row];
        float rnB = rnorms[(p ? 2 : 3) * NN + row];
        (p ? dt : dv)[row] = d * rnA * rnB;
    }
}

// ---------------------------------------------------------------- fused WMMA GEMM + streaming logsumexp/rank
// LDS map (exactly 64KB):
//   [0,32768)        A tile : 32 rows x 512 K f16, XOR-swizzled 16B granules (key = row&15)
//   [32768,49152)    B buf0 : 64 cols x 128 K f16, XOR-swizzled (key = col&15)
//   [49152,65536)    B buf1
//   stats scratch (1.5KB) aliases B buf0 between gemms (fenced by barriers)
__global__ __launch_bounds__(256)
void fused_gemm_kernel(const _Float16* __restrict__ vn16, const _Float16* __restrict__ tn16,
                       const _Float16* __restrict__ pvn16, const _Float16* __restrict__ ptn16,
                       const float* __restrict__ dv, const float* __restrict__ dt,
                       float* __restrict__ denom, int* __restrict__ posv, int* __restrict__ post) {
    __shared__ __align__(16) char smem[65536];
    _Float16* Al  = (_Float16*)smem;                  // 32*512
    _Float16* Bl0 = (_Float16*)(smem + 32768);        // 64*128
    _Float16* Bl1 = (_Float16*)(smem + 49152);        // 64*128
    float* sM = (float*)(smem + 32768);               // [4][32]  (aliases Bl0)
    float* sS = sM + 128;                             // [4][32]
    int*   sC = (int*)(sS + 128);                     // [4][32]

    const int tid  = threadIdx.x;
    const int lane = tid & 31;
    const int wave = tid >> 5;
    const int wm   = wave >> 2;      // 0..1 : 16-row group
    const int wn   = wave & 3;       // 0..3 : 16-col group
    const int half = lane >> 4;      // VGPR-layout half
    const int l16  = lane & 15;
    const int rb   = blockIdx.x * 32;

    const _Float16* Amats[2] = { vn16, tn16 };
    const _Float16* Bmats[2] = { ptn16, pvn16 };

    float gM[2] = {0.f, 0.f}, gS[2] = {0.f, 0.f}; int gC[2] = {0, 0};

    for (int g = 0; g < 2; ++g) {
        const _Float16* Ag = Amats[g];
        const _Float16* Bg = Bmats[g];
        const float*    Dg = g ? dt : dv;
        __syncthreads();   // all waves done with previous gemm's LDS (incl. stats reads)

        // async-stage A tile: 2048 swizzled 16B granules, 8 per thread
        #pragma unroll
        for (int i = 0; i < 8; ++i) {
            int id = tid + i * 256;
            int r = id >> 6, gi = id & 63;
            asyncCopyB128(ldsOff(&Al[r * 512 + ((gi ^ (r & 15)) * 8)]), Ag,
                          (uint32_t)((((rb + r) * CC) + gi * 8) * 2));
        }
        // async-stage B chunk for stage 0 into buffer 0: 1024 granules, 4 per thread
        #pragma unroll
        for (int i = 0; i < 4; ++i) {
            int id = tid + i * 256;
            int c = id >> 4, gb = id & 15;
            asyncCopyB128(ldsOff(&Bl0[c * 128 + ((gb ^ (c & 15)) * 8)]), Bg,
                          (uint32_t)((c * CC + gb * 8) * 2));
        }

        float m[8], s[8], dvl[8]; int cnt[8];
        #pragma unroll
        for (int r = 0; r < 8; ++r) {
            m[r] = -__builtin_inff(); s[r] = 0.f; cnt[r] = 0;
            dvl[r] = Dg[rb + wm * 16 + half * 8 + r];
        }

        const int rowA = wm * 16 + l16;
        const int colB = wn * 16 + l16;

        for (int nit = 0; nit < 64; ++nit) {
            v8f acc = {};
            for (int kc = 0; kc < NKC; ++kc) {
                const int q   = nit * NKC + kc;
                const int cur = q & 1;
                waitAsync0();        // batch q landed in LDS
                __syncthreads();     // visible to all waves; all waves done reading other buffer
                if (q + 1 < NSTAGE) {
                    const int nb2 = ((q + 1) >> 2) * 64;
                    const int kc2 = (q + 1) & 3;
                    _Float16* Bnxt = cur ? Bl0 : Bl1;
                    #pragma unroll
                    for (int i = 0; i < 4; ++i) {
                        int id = tid + i * 256;
                        int c = id >> 4, gb = id & 15;
                        asyncCopyB128(ldsOff(&Bnxt[c * 128 + ((gb ^ (c & 15)) * 8)]), Bg,
                                      (uint32_t)((((nb2 + c) * CC) + kc2 * 128 + gb * 8) * 2));
                    }
                }
                const _Float16* Bcur = cur ? Bl1 : Bl0;
                #pragma unroll
                for (int ks = 0; ks < 4; ++ks) {
                    // A frag (16x32): lane holds M=l16, K = kb + 16*(e>=8) + 8*half + (e&7)
                    const int giA = kc * 16 + ks * 4 + half;          // lo granule
                    V16 a;
                    a.h[0] = *(const v8h*)&Al[rowA * 512 + ((giA       ^ l16) * 8)];
                    a.h[1] = *(const v8h*)&Al[rowA * 512 + (((giA + 2) ^ l16) * 8)];
                    // B frag (32x16): lane holds N=l16, K = ks*32 + 16*(e>=8) + 8*half + (e&7)
                    const int gbB = ks * 4 + half;
                    V16 b;
                    b.h[0] = *(const v8h*)&Bcur[colB * 128 + ((gbB       ^ l16) * 8)];
                    b.h[1] = *(const v8h*)&Bcur[colB * 128 + (((gbB + 2) ^ l16) * 8)];
                    acc = __builtin_amdgcn_wmma_f32_16x16x32_f16(
                        false, a.v, false, b.v, (short)0, acc, false, false);
                }
            }
            // streaming per-row stats (lane owns one column of 8 rows)
            const int colg = nit * 64 + wn * 16 + l16;
            #pragma unroll
            for (int r = 0; r < 8; ++r) {
                float x = acc[r];
                int Rrow = rb + wm * 16 + half * 8 + r;
                cnt[r] += ((x > dvl[r]) && (colg != Rrow)) ? 1 : 0;
                float xs = x * 10.f;                       // 1/TEMPERATURE
                float nm = fmaxf(m[r], xs);
                s[r] = s[r] * __expf(m[r] - nm) + __expf(xs - nm);
                m[r] = nm;
            }
        }
        // merge across the 16 lanes holding each row, write per-wave partials
        #pragma unroll
        for (int r = 0; r < 8; ++r) {
            float mm = m[r], ssv = s[r]; int cc = cnt[r];
            #pragma unroll
            for (int off = 1; off < 16; off <<= 1) {
                float om = __shfl_xor(mm, off, 32);
                float os = __shfl_xor(ssv, off, 32);
                int   oc = __shfl_xor(cc, off, 32);
                float nm = fmaxf(mm, om);
                ssv = ssv * __expf(mm - nm) + os * __expf(om - nm);
                mm = nm; cc += oc;
            }
            if (l16 == 0) {
                int rl = wm * 16 + half * 8 + r;
                sM[wn * 32 + rl] = mm; sS[wn * 32 + rl] = ssv; sC[wn * 32 + rl] = cc;
            }
        }
        __syncthreads();
        if (tid < 32) {   // fold the 4 wn partials into registers per gemm
            float M = -__builtin_inff(), S = 0.f; int cc = 0;
            #pragma unroll
            for (int w = 0; w < 4; ++w) {
                float mm = sM[w * 32 + tid], ssv = sS[w * 32 + tid];
                float nm = fmaxf(M, mm);
                S = S * __expf(M - nm) + ssv * __expf(mm - nm);
                M = nm;
                cc += sC[w * 32 + tid];
            }
            gM[g] = M; gS[g] = S; gC[g] = cc;
        }
    }
    if (tid < 32) {
        float M = fmaxf(gM[0], gM[1]);
        float S = gS[0] * __expf(gM[0] - M) + gS[1] * __expf(gM[1] - M);
        denom[rb + tid] = M + __logf(S);
        posv[rb + tid] = gC[0];
        post[rb + tid] = gC[1];
    }
}

// ---------------------------------------------------------------- final scalar reductions
__device__ float blockSum(float v, float* red) {
    int tid = threadIdx.x;
    __syncthreads();
    red[tid] = v; __syncthreads();
    for (int o = 128; o > 0; o >>= 1) { if (tid < o) red[tid] += red[tid + o]; __syncthreads(); }
    return red[0];
}
__device__ float blockMax(float v, float* red) {
    int tid = threadIdx.x;
    __syncthreads();
    red[tid] = v; __syncthreads();
    for (int o = 128; o > 0; o >>= 1) { if (tid < o) red[tid] = fmaxf(red[tid], red[tid + o]); __syncthreads(); }
    return red[0];
}

__global__ __launch_bounds__(256)
void finalize_kernel(const float* __restrict__ dv, const float* __restrict__ dt,
                     const float* __restrict__ denom,
                     const int* __restrict__ posv, const int* __restrict__ post,
                     const float* __restrict__ colsum, const float* __restrict__ colsq,
                     float* __restrict__ out) {
    __shared__ float red[256];
    const int tid = threadIdx.x;

    // nominator: logsumexp over both scaled diagonals (2N values)
    float lm = -__builtin_inff();
    for (int i = tid; i < 2 * NN; i += 256) {
        float d = ((i < NN) ? dv[i] : dt[i - NN]) * 10.f;
        lm = fmaxf(lm, d);
    }
    float M = blockMax(lm, red);
    float le = 0.f;
    for (int i = tid; i < 2 * NN; i += 256) {
        float d = ((i < NN) ? dv[i] : dt[i - NN]) * 10.f;
        le += __expf(d - M);
    }
    float nom = M + __logf(blockSum(le, red));

    float sd = 0.f;
    for (int i = tid; i < NN; i += 256) sd += denom[i];
    float loss = blockSum(sd, red) / (float)NN - nom;

    float a1 = 0, a5 = 0, a10 = 0, am = 0, b1 = 0, b5 = 0, b10 = 0, bm = 0;
    for (int i = tid; i < NN; i += 256) {
        int pv_ = posv[i], pt_ = post[i];
        a1 += (pv_ < 1); a5 += (pv_ < 5); a10 += (pv_ < 10); am += (float)pv_;
        b1 += (pt_ < 1); b5 += (pt_ < 5); b10 += (pt_ < 10); bm += (float)pt_;
    }
    float r[8];
    r[0] = blockSum(a1, red); r[1] = blockSum(a5, red); r[2] = blockSum(a10, red); r[3] = blockSum(am, red);
    r[4] = blockSum(b1, red); r[5] = blockSum(b5, red); r[6] = blockSum(b10, red); r[7] = blockSum(bm, red);

    float stds[4];
    for (int mat = 0; mat < 4; ++mat) {
        float loc = 0.f;
        for (int c = tid; c < CC; c += 256) {
            float s1 = colsum[mat * CC + c];
            float s2 = colsq[mat * CC + c];
            float var = (s2 - s1 * s1 / (float)NN) / (float)(NN - 1);
            loc += sqrtf(fmaxf(var, 0.f));
        }
        stds[mat] = blockSum(loc, red) / (float)CC;
    }

    if (tid == 0) {
        out[0] = loss;
        out[1] = stds[0]; out[2] = stds[1]; out[3] = stds[2]; out[4] = stds[3];
        out[5] = r[0] / NN; out[6] = r[1] / NN; out[7] = r[2] / NN; out[8]  = r[3] / NN;
        out[9] = r[4] / NN; out[10] = r[5] / NN; out[11] = r[6] / NN; out[12] = r[7] / NN;
    }
}

// ---------------------------------------------------------------- launch
extern "C" void kernel_launch(void* const* d_in, const int* in_sizes, int n_in,
                              void* d_out, int out_size, void* d_ws, size_t ws_size,
                              hipStream_t stream) {
    (void)in_sizes; (void)n_in; (void)out_size; (void)ws_size;
    const float* v  = (const float*)d_in[0];
    const float* t  = (const float*)d_in[1];
    const float* pv = (const float*)d_in[2];
    const float* pt = (const float*)d_in[3];
    float* out = (float*)d_out;

    char* p = (char*)d_ws;
    const size_t matB = (size_t)NN * CC * sizeof(_Float16);
    _Float16* nm16 = (_Float16*)p; p += 4 * matB;            // vn,tn,pvn,ptn (f16)
    float* colsum = (float*)p; p += 4 * CC * sizeof(float);
    float* colsq  = (float*)p; p += 4 * CC * sizeof(float);
    float* rnorms = (float*)p; p += 4 * NN * sizeof(float);
    float* dv     = (float*)p; p += NN * sizeof(float);
    float* dt     = (float*)p; p += NN * sizeof(float);
    float* denom  = (float*)p; p += NN * sizeof(float);
    int*   posv   = (int*)p;   p += NN * sizeof(int);
    int*   post   = (int*)p;   p += NN * sizeof(int);

    zero_kernel<<<(4 * CC * 2 + 255) / 256, 256, 0, stream>>>(colsum, 4 * CC * 2);
    normalize_kernel<<<dim3(NN / 8, 4), 256, 0, stream>>>(v, t, pv, pt, nm16, rnorms, colsum, colsq);
    diag_kernel<<<dim3(NN / 8, 2), 256, 0, stream>>>(v, t, pv, pt, rnorms, dv, dt);
    fused_gemm_kernel<<<NN / 32, 256, 0, stream>>>(
        nm16 + 0 * (size_t)NN * CC,   // vn
        nm16 + 1 * (size_t)NN * CC,   // tn
        nm16 + 2 * (size_t)NN * CC,   // pvn
        nm16 + 3 * (size_t)NN * CC,   // ptn
        dv, dt, denom, posv, post);
    finalize_kernel<<<1, 256, 0, stream>>>(dv, dt, denom, posv, post, colsum, colsq, out);
}
